// EntityGraphRefinement_7696581394665
// MI455X (gfx1250) — matrix operand order repaired and checked by reading, not compiled, not run on hardware
//
#include <hip/hip_runtime.h>
#include <hip/hip_bf16.h>

typedef __attribute__((ext_vector_type(16))) _Float16 v16h;
typedef __attribute__((ext_vector_type(8)))  float    v8f;
typedef __attribute__((ext_vector_type(2)))  __fp16   v2fp16;

#define N_ENT 128
#define E_DIM 768
#define D_EMB 768
#define HEADS 12
#define KEY_DIM 64
#define BATCH 4
#define SEQ 1024

__device__ __forceinline__ float gelu_tanh(float x) {
    float x3 = x * x * x;
    return 0.5f * x * (1.0f + tanhf(0.7978845608028654f * (x + 0.044715f * x3)));
}

// single v_cvt_pk_rz_f16_f32: pack two f32 into one dword of f16 pairs
__device__ __forceinline__ unsigned pack_f16x2(float a, float b) {
    union { v2fp16 h; unsigned u; } t;
    t.h = __builtin_amdgcn_cvt_pkrtz(a, b);
    return t.u;
}

union V16U { uint4 u[2]; v16h h; };

// ---------------------------------------------------------------------------
// Stage one K-step's global tiles into registers (f32, unconverted).
// A: 8 float2 per thread (128x32-half tile), B: 4 float2 per thread (64x32).
// ---------------------------------------------------------------------------
template<bool HAS_ADD, bool GUARD_M>
__device__ __forceinline__ void stage_load(
    const float* __restrict__ A, const float* __restrict__ B,
    const float* __restrict__ addv, int addmod,
    int M, int K, int ldb, int row0, int col0, int t, int k0,
    float2* aR, float2* bR)
{
#pragma unroll
    for (int it = 0; it < 8; ++it) {
        int idx = t + it * 256;
        int r = idx >> 4, pc = idx & 15;
        int gr = row0 + r, gc = k0 + 2 * pc;
        float2 a = make_float2(0.0f, 0.0f);
        if (!GUARD_M || gr < M) {
            a = *(const float2*)(A + (size_t)gr * K + gc);
            if (HAS_ADD) {
                float2 v = *(const float2*)(addv + (size_t)(gr % addmod) * K + gc);
                a.x += v.x; a.y += v.y;
            }
        }
        aR[it] = a;
    }
#pragma unroll
    for (int it = 0; it < 4; ++it) {
        int idx = t + it * 256;
        int n = idx & 63, kp = idx >> 6;
        bR[it].x = B[(size_t)(k0 + 2 * kp) * ldb + col0 + n];
        bR[it].y = B[(size_t)(k0 + 2 * kp + 1) * ldb + col0 + n];
    }
}

// ---------------------------------------------------------------------------
// Tiled f16-WMMA GEMM: C[M,N] = (A[M,K] (+ addv bcast rows)) * B[K,N] (+bias)
// Requires K % 32 == 0, N % 64 == 0; GUARD_M=false additionally requires
// M % 128 == 0. 256 threads = 8 waves; macro tile 128x64; wave: 16 x 64.
// LDS tiles are packed f16-pair dwords; B stored transposed (n-major) so
// every WMMA fragment is two aligned ds_load_b128. K-loop is register-
// prefetch software-pipelined to overlap global loads with DS+WMMA.
// ---------------------------------------------------------------------------
template<bool HAS_ADD, bool GUARD_M>
__global__ __launch_bounds__(256) void gemm_wmma_f16(
    const float* __restrict__ A, const float* __restrict__ B,
    const float* __restrict__ bias, float* __restrict__ C,
    int M, int N, int K, int ldb,
    const float* __restrict__ addv, int addmod)
{
    __shared__ unsigned As32[128][20];  // 128 rows x (16 dwords = 32 halfs) + pad
    __shared__ unsigned Bs32[64][20];   // 64 n-rows x (16 dwords = 32 k-halfs) + pad

    const int t    = threadIdx.x;
    const int wave = t >> 5;
    const int lane = t & 31;
    const int row0 = blockIdx.y * 128;
    const int col0 = blockIdx.x * 64;
    const int m0   = wave * 16;

    v8f acc[4] = {};
    float2 aCur[8], bCur[4], aNxt[8], bNxt[4];

    stage_load<HAS_ADD, GUARD_M>(A, B, addv, addmod, M, K, ldb,
                                 row0, col0, t, 0, aCur, bCur);

    for (int k0 = 0; k0 < K; k0 += 32) {
        // write current tile (f32 -> packed f16x2) to LDS
#pragma unroll
        for (int it = 0; it < 8; ++it) {
            int idx = t + it * 256;
            As32[idx >> 4][idx & 15] = pack_f16x2(aCur[it].x, aCur[it].y);
        }
#pragma unroll
        for (int it = 0; it < 4; ++it) {
            int idx = t + it * 256;
            Bs32[idx & 63][idx >> 6] = pack_f16x2(bCur[it].x, bCur[it].y);
        }
        __syncthreads();

        // prefetch next tile's global data while WMMA works on this one
        if (k0 + 32 < K) {
            stage_load<HAS_ADD, GUARD_M>(A, B, addv, addmod, M, K, ldb,
                                         row0, col0, t, k0 + 32, aNxt, bNxt);
        }

        // A fragment: lanes 0-15 -> K 0..7 / 16..23 ; lanes 16-31 -> K 8..15 / 24..31
        const int mrow = m0 + (lane & 15);
        const int kha  = (lane >> 4) * 4;      // dword offset: 0 or 4
        V16U av;
        av.u[0] = *(const uint4*)&As32[mrow][kha];
        av.u[1] = *(const uint4*)&As32[mrow][8 + kha];

        const int nc  = lane & 15;
        const int kbd = (lane >> 4) * 4;       // dword offset: 0 or 4
#pragma unroll
        for (int f = 0; f < 4; ++f) {
            V16U bv;
            bv.u[0] = *(const uint4*)&Bs32[f * 16 + nc][kbd];
            bv.u[1] = *(const uint4*)&Bs32[f * 16 + nc][8 + kbd];
            acc[f] = __builtin_amdgcn_wmma_f32_16x16x32_f16(
                false, av.h, false, bv.h, (short)0, acc[f], false, false);
        }
        __syncthreads();

#pragma unroll
        for (int it = 0; it < 8; ++it) aCur[it] = aNxt[it];
#pragma unroll
        for (int it = 0; it < 4; ++it) bCur[it] = bNxt[it];
    }

    // epilogue: C/D layout — lane&15 = N, lane>>4 selects M half, vgpr r -> M = r + 8*half
    const int ncE = lane & 15;
    const int mh  = lane >> 4;
#pragma unroll
    for (int f = 0; f < 4; ++f) {
        int gc = col0 + f * 16 + ncE;
        float bv = bias ? bias[gc] : 0.0f;
#pragma unroll
        for (int r = 0; r < 8; ++r) {
            int gr = row0 + m0 + r + 8 * mh;
            if (!GUARD_M || gr < M) C[(size_t)gr * N + gc] = acc[f][r] + bv;
        }
    }
}

// ---------------------------------------------------------------------------
// context[b,d] = mean_s (emb[b,s,d] + pos[s,d])
// ---------------------------------------------------------------------------
__global__ __launch_bounds__(256) void context_kernel(
    const float* __restrict__ emb, const float* __restrict__ pos,
    float* __restrict__ ctxmean)
{
    int d = blockIdx.x * 256 + threadIdx.x;
    int b = blockIdx.y;
    if (d >= D_EMB) return;
    float s = 0.0f;
    for (int si = 0; si < SEQ; ++si)
        s += emb[((size_t)(b * SEQ + si)) * D_EMB + d] + pos[(size_t)si * D_EMB + d];
    ctxmean[b * D_EMB + d] = s * (1.0f / (float)SEQ);
}

// ---------------------------------------------------------------------------
// Attention: per (n,h,b) block. scores -> softmax -> ctx ; inv_denom = max attn
// ---------------------------------------------------------------------------
__global__ __launch_bounds__(256) void attn_kernel(
    const float* __restrict__ q,    // (N, H*K)
    const float* __restrict__ kbuf, // (B*S, H*K)
    const float* __restrict__ vbuf, // (B*S, H*K)
    float* __restrict__ ctx,        // (B*N, H*K)
    float* __restrict__ invd)       // (B, H, N)
{
    const int n = blockIdx.x, h = blockIdx.y, b = blockIdx.z;
    __shared__ float qs[KEY_DIM];
    __shared__ float p[SEQ];
    __shared__ float red[256];
    __shared__ float part[4][KEY_DIM];
    const int t = threadIdx.x;

    if (t < KEY_DIM) qs[t] = q[(size_t)n * (HEADS * KEY_DIM) + h * KEY_DIM + t];
    __syncthreads();

    float sc[4];
    float lmax = -3.4e38f;
#pragma unroll
    for (int c = 0; c < 4; ++c) {
        int s = t + c * 256;
        const float* krow = kbuf + ((size_t)(b * SEQ + s)) * (HEADS * KEY_DIM) + h * KEY_DIM;
        float acc = 0.0f;
#pragma unroll
        for (int kk = 0; kk < KEY_DIM; kk += 4) {
            float4 kv = *(const float4*)(krow + kk);
            acc += qs[kk] * kv.x + qs[kk + 1] * kv.y + qs[kk + 2] * kv.z + qs[kk + 3] * kv.w;
        }
        acc *= 0.125f;  // 1/sqrt(64)
        sc[c] = acc;
        lmax = fmaxf(lmax, acc);
    }
    red[t] = lmax; __syncthreads();
    for (int off = 128; off; off >>= 1) {
        if (t < off) red[t] = fmaxf(red[t], red[t + off]);
        __syncthreads();
    }
    const float m = red[0];
    __syncthreads();

    float lsum = 0.0f;
#pragma unroll
    for (int c = 0; c < 4; ++c) {
        float e = __expf(sc[c] - m);
        p[t + c * 256] = e;
        lsum += e;
    }
    red[t] = lsum; __syncthreads();
    for (int off = 128; off; off >>= 1) {
        if (t < off) red[t] += red[t + off];
        __syncthreads();
    }
    const float inv = 1.0f / red[0];   // == max_s softmax(scores)[s]
    if (t == 0) invd[(b * HEADS + h) * N_ENT + n] = inv;

    // ctx[kk] = inv * sum_s p[s] * v[b,s,h,kk]
    const int kk = t & 63, c = t >> 6;
    float acc = 0.0f;
    for (int s = c * 256; s < c * 256 + 256; ++s)
        acc += p[s] * vbuf[((size_t)(b * SEQ + s)) * (HEADS * KEY_DIM) + h * KEY_DIM + kk];
    part[c][kk] = acc; __syncthreads();
    if (c == 0) {
        float tot = (part[0][kk] + part[1][kk] + part[2][kk] + part[3][kk]) * inv;
        ctx[((size_t)(b * N_ENT + n)) * (HEADS * KEY_DIM) + h * KEY_DIM + kk] = tot;
    }
}

// ---------------------------------------------------------------------------
// mask[b,n] = (max_h inv_denom > 1e-3)
// ---------------------------------------------------------------------------
__global__ __launch_bounds__(256) void mask_kernel(
    const float* __restrict__ invd, float* __restrict__ mask_out,
    float* __restrict__ mask_ws)
{
    int idx = blockIdx.x * 256 + threadIdx.x;
    if (idx >= BATCH * N_ENT) return;
    int b = idx >> 7, n = idx & 127;
    float a = 0.0f;
    for (int h = 0; h < HEADS; ++h)
        a = fmaxf(a, invd[(b * HEADS + h) * N_ENT + n]);
    float mk = (a > 0.001f) ? 1.0f : 0.0f;
    mask_out[idx] = mk;
    mask_ws[idx]  = mk;
}

// ---------------------------------------------------------------------------
// Per-edge kernel: one wave32 per edge; 24 features/lane held in VGPRs.
// ---------------------------------------------------------------------------
__global__ __launch_bounds__(256) void edge_kernel(
    const float* __restrict__ Ai,  const float* __restrict__ Bj,
    const float* __restrict__ Cc,  const float* __restrict__ w1g,
    const float* __restrict__ b1,  const float* __restrict__ W2,
    const float* __restrict__ b2,
    const float* __restrict__ Ag,  const float* __restrict__ Bg,
    const float* __restrict__ bg1, const float* __restrict__ Wg2,
    const float* __restrict__ bg2,
    const float* __restrict__ graph_init, const float* __restrict__ mask,
    float* __restrict__ graph_out)
{
    const int wave = threadIdx.x >> 5;
    const int lane = threadIdx.x & 31;
    const int edge = blockIdx.x * 8 + wave;
    const int b = edge >> 14, rem = edge & 16383, i = rem >> 7, j = rem & 127;

    const float* ai = Ai + (size_t)(b * N_ENT + i) * E_DIM;
    const float* bj = Bj + (size_t)(b * N_ENT + j) * E_DIM;
    const float* cc = Cc + (size_t)b * E_DIM;

    float s[24], wg[24], w2r[24];
#pragma unroll
    for (int ti = 0; ti < 24; ++ti) {
        int e = lane + 32 * ti;
        s[ti]   = ai[e] + bj[e] + cc[e] + b1[e];
        wg[ti]  = w1g[e];
        w2r[ti] = W2[e];
    }

    float g = graph_init[edge] * 1.6f - 0.8f;
    const float b2v = b2[0];

#pragma unroll
    for (int step = 0; step < 3; ++step) {
        float acc = 0.0f;
#pragma unroll
        for (int ti = 0; ti < 24; ++ti) {
            float hx = s[ti] + g * wg[ti];
            acc += gelu_tanh(hx) * w2r[ti];
        }
#pragma unroll
        for (int off = 16; off; off >>= 1) acc += __shfl_xor(acc, off, 32);
        g += tanhf(acc + b2v);
    }

    // sparsification gate (independent of g)
    const float* agi = Ag + (size_t)(b * N_ENT + i) * E_DIM;
    const float* bgj = Bg + (size_t)(b * N_ENT + j) * E_DIM;
    float gacc = 0.0f;
#pragma unroll
    for (int ti = 0; ti < 24; ++ti) {
        int e = lane + 32 * ti;
        float hx = agi[e] + bgj[e] + bg1[e];
        gacc += gelu_tanh(hx) * Wg2[e];
    }
#pragma unroll
    for (int off = 16; off; off >>= 1) gacc += __shfl_xor(gacc, off, 32);
    float gate = 1.0f / (1.0f + __expf(-(gacc + bg2[0])));

    g *= gate * mask[b * N_ENT + i] * mask[b * N_ENT + j];
    graph_out[edge] = g;
}

// ---------------------------------------------------------------------------
// entities_out = entities * mask[b,n]
// ---------------------------------------------------------------------------
__global__ __launch_bounds__(256) void apply_mask_kernel(
    const float* __restrict__ ent, const float* __restrict__ mask,
    float* __restrict__ out, int total)
{
    int idx = blockIdx.x * 256 + threadIdx.x;
    if (idx >= total) return;
    out[idx] = ent[idx] * mask[idx / E_DIM];
}

// ---------------------------------------------------------------------------
extern "C" void kernel_launch(void* const* d_in, const int* in_sizes, int n_in,
                              void* d_out, int out_size, void* d_ws, size_t ws_size,
                              hipStream_t stream)
{
    const float* emb   = (const float*)d_in[0];   // (B,S,D)
    const float* ginit = (const float*)d_in[1];   // (B,N,N)
    const float* elib  = (const float*)d_in[2];   // (N,E)
    const float* pos   = (const float*)d_in[3];   // (MAX_SEQ,D)
    const float* Wq    = (const float*)d_in[4];
    const float* bq    = (const float*)d_in[5];
    const float* Wk    = (const float*)d_in[6];
    const float* bk    = (const float*)d_in[7];
    const float* Wv    = (const float*)d_in[8];
    const float* bv    = (const float*)d_in[9];
    const float* Wo    = (const float*)d_in[10];
    const float* bo    = (const float*)d_in[11];
    const float* W1    = (const float*)d_in[12];  // (2305,768)
    const float* b1    = (const float*)d_in[13];
    const float* W2    = (const float*)d_in[14];  // (768,1)
    const float* b2    = (const float*)d_in[15];
    const float* Wg1   = (const float*)d_in[16];  // (1536,768)
    const float* bg1   = (const float*)d_in[17];
    const float* Wg2   = (const float*)d_in[18];  // (768,1)
    const float* bg2   = (const float*)d_in[19];

    float* ws = (float*)d_ws;
    size_t o = 0;
    float* kb      = ws + o; o += (size_t)BATCH * SEQ * 768;
    float* vb      = ws + o; o += (size_t)BATCH * SEQ * 768;
    float* qb      = ws + o; o += (size_t)N_ENT * 768;
    float* ctxa    = ws + o; o += (size_t)BATCH * N_ENT * 768;
    float* invd    = ws + o; o += (size_t)BATCH * HEADS * N_ENT;
    float* ent     = ws + o; o += (size_t)BATCH * N_ENT * 768;
    float* fAi     = ws + o; o += (size_t)BATCH * N_ENT * 768;
    float* fBj     = ws + o; o += (size_t)BATCH * N_ENT * 768;
    float* fAg     = ws + o; o += (size_t)BATCH * N_ENT * 768;
    float* fBg     = ws + o; o += (size_t)BATCH * N_ENT * 768;
    float* ctxmean = ws + o; o += (size_t)BATCH * D_EMB;
    float* fCc     = ws + o; o += (size_t)BATCH * E_DIM;
    float* maskw   = ws + o; o += (size_t)BATCH * N_ENT;

    float* out       = (float*)d_out;
    float* ent_out   = out;                                  // (B,N,E)
    float* graph_out = out + (size_t)BATCH * N_ENT * E_DIM;  // (B,N,N)
    float* mask_out  = graph_out + (size_t)BATCH * N_ENT * N_ENT;

    // 1) global context (fused emb+pos)
    context_kernel<<<dim3(3, BATCH), 256, 0, stream>>>(emb, pos, ctxmean);

    // 2) k/v projections with fused positional add: M = B*S = 4096
    gemm_wmma_f16<true, false><<<dim3(12, 32), 256, 0, stream>>>(
        emb, Wk, bk, kb, BATCH * SEQ, 768, 768, 768, pos, SEQ);
    gemm_wmma_f16<true, false><<<dim3(12, 32), 256, 0, stream>>>(
        emb, Wv, bv, vb, BATCH * SEQ, 768, 768, 768, pos, SEQ);

    // 3) q projection: M = 128
    gemm_wmma_f16<false, false><<<dim3(12, 1), 256, 0, stream>>>(
        elib, Wq, bq, qb, N_ENT, 768, 768, 768, nullptr, 1);

    // 4) attention + activity
    attn_kernel<<<dim3(N_ENT, HEADS, BATCH), 256, 0, stream>>>(qb, kb, vb, ctxa, invd);
    mask_kernel<<<dim3(2), 256, 0, stream>>>(invd, mask_out, maskw);

    // 5) entity output projection: M = B*N = 512
    gemm_wmma_f16<false, false><<<dim3(12, 4), 256, 0, stream>>>(
        ctxa, Wo, bo, ent, BATCH * N_ENT, 768, 768, 768, nullptr, 1);

    // 6) factorized refinement / gate precomputes
    gemm_wmma_f16<false, false><<<dim3(12, 4), 256, 0, stream>>>(
        ent, W1, nullptr, fAi, BATCH * N_ENT, 768, 768, 768, nullptr, 1);
    gemm_wmma_f16<false, false><<<dim3(12, 4), 256, 0, stream>>>(
        ent, W1 + (size_t)768 * 768, nullptr, fBj, BATCH * N_ENT, 768, 768, 768, nullptr, 1);
    gemm_wmma_f16<false, true><<<dim3(12, 1), 256, 0, stream>>>(
        ctxmean, W1 + (size_t)1537 * 768, nullptr, fCc, BATCH, 768, 768, 768, nullptr, 1);
    gemm_wmma_f16<false, false><<<dim3(12, 4), 256, 0, stream>>>(
        ent, Wg1, nullptr, fAg, BATCH * N_ENT, 768, 768, 768, nullptr, 1);
    gemm_wmma_f16<false, false><<<dim3(12, 4), 256, 0, stream>>>(
        ent, Wg1 + (size_t)768 * 768, nullptr, fBg, BATCH * N_ENT, 768, 768, 768, nullptr, 1);

    // 7) per-edge refinement + gate + mask: 65536 edges, 8 waves/block
    const float* w1g = W1 + (size_t)1536 * 768;
    edge_kernel<<<dim3(BATCH * N_ENT * N_ENT / 8), 256, 0, stream>>>(
        fAi, fBj, fCc, w1g, b1, W2, b2, fAg, fBg, bg1, Wg2, bg2,
        ginit, maskw, graph_out);

    // 8) masked entities to output
    apply_mask_kernel<<<dim3((BATCH * N_ENT * E_DIM + 255) / 256), 256, 0, stream>>>(
        ent, maskw, ent_out, BATCH * N_ENT * E_DIM);
}
